// VQVAE_22273700397838
// MI455X (gfx1250) — compile-verified, matrix operand order checked by reading, not run on hardware
//
#include <hip/hip_runtime.h>
#include <hip/hip_bf16.h>

// ---------------------------------------------------------------------------
// VQ-VAE forward for MI455X (gfx1250, wave32)
//   - 4 big GEMMs on v_wmma_f32_16x16x32_f16 (BM=128 BN=128 BK=32, 8 waves)
//   - f16-input GEMMs use double-buffered LDS fed by
//     global_load_async_to_lds_b128 (ASYNCcnt), 1 barrier / K-step
//   - layer-1 GEMM (f32 input, K=784) uses sync staging, with the K-tail
//     guard hoisted out of the main loop (24 clean tiles + 1 guarded tile)
//   - tiny middle (256->10, VQ argmin over 64 codes, 10->256) fused per-row
// Outputs (flat, f32): xp[B*784] | z_e[B*10] | z_q[B*10]
// ---------------------------------------------------------------------------

typedef _Float16 v16h __attribute__((ext_vector_type(16)));
typedef _Float16 v8h  __attribute__((ext_vector_type(8)));
typedef float    v8f  __attribute__((ext_vector_type(8)));
typedef float    v4f  __attribute__((ext_vector_type(4)));

#define DIM_IN 784
#define DIM1   512
#define DIM2   256
#define DIMZ   10
#define VOCAB  64

// ---------------------------------------------------------------------------
// CDNA5 async copy: global -> LDS, 16 bytes per lane, tracked by ASYNCcnt.
// GV mode: lds-offset VGPR, 64-bit global address VGPR pair, no saddr.
// ---------------------------------------------------------------------------
__device__ __forceinline__ void async_copy_b128(unsigned int lds_off,
                                                unsigned long long gaddr) {
  asm volatile("global_load_async_to_lds_b128 %0, %1, off"
               :
               : "v"(lds_off), "v"(gaddr)
               : "memory");
}
__device__ __forceinline__ void wait_async0() {
  asm volatile("s_wait_asynccnt 0x0" ::: "memory");
}
// Low 32 bits of a generic LDS pointer == wave-relative LDS byte address.
__device__ __forceinline__ unsigned int lds_addr32(const void* p) {
  return (unsigned int)(unsigned long long)(uintptr_t)p;
}

// ---------------------------------------------------------------------------
// Weight convert + transpose: W[K,N] f32  ->  Wt[N,K] f16 (row-major in K)
// ---------------------------------------------------------------------------
__global__ __launch_bounds__(256) void wconv_kernel(
    const float* __restrict__ W, _Float16* __restrict__ Wt, int K, int N) {
  int i = blockIdx.x * 256 + threadIdx.x;
  if (i >= K * N) return;
  int n = i / K;
  int k = i - n * K;
  Wt[i] = (_Float16)W[(size_t)k * N + n];
}

// ---------------------------------------------------------------------------
// WMMA fragment load (16-bit A/B layout): lane l holds row/col (l&15),
// halves K = [kb..kb+7] ++ [kb+16..kb+23], kb = (l>>4)*8.
// ---------------------------------------------------------------------------
__device__ __forceinline__ v16h load_frag(const _Float16* __restrict__ base, int kb) {
  v8h lo = *(const v8h*)(base + kb);
  v8h hi = *(const v8h*)(base + kb + 16);
  return __builtin_shufflevector(lo, hi, 0, 1, 2, 3, 4, 5, 6, 7,
                                         8, 9, 10, 11, 12, 13, 14, 15);
}

// ---------------------------------------------------------------------------
// Tiled WMMA GEMM:  Out = act(A[M,K] @ Wt^T + bias)   (Wt stored [N,K] f16)
// MODE 0: A f32 (convert in staging, sync path, zero-pads K tail), relu, f16 out
// MODE 1: A f16, async double-buffered staging, relu,    f16 out
// MODE 2: A f16, async double-buffered staging, sigmoid, f32 out
// ---------------------------------------------------------------------------
template <int MODE>
__global__ __launch_bounds__(256) void gemm_wmma_kernel(
    const void* __restrict__ Aptr, const _Float16* __restrict__ Wt,
    const float* __restrict__ bias, void* __restrict__ Out,
    int N, int K) {
  constexpr int BM = 128, BN = 128, BK = 32, LDT = 40;  // 40-half row pad
  constexpr int NBUF = (MODE == 0) ? 1 : 2;
  __shared__ _Float16 As[NBUF * BM * LDT];
  __shared__ _Float16 Bs[NBUF * BN * LDT];

  const int t    = threadIdx.x;
  const int wave = t >> 5;
  const int lane = t & 31;
  const int wm   = wave >> 1;  // 0..3  -> 32-row strip
  const int wn   = wave & 1;   // 0..1  -> 64-col strip
  const int m0   = blockIdx.y * BM;
  const int n0   = blockIdx.x * BN;

  v8f acc[2][4] = {};

  const int li = lane & 15;
  const int kb = (lane >> 4) * 8;
  const int KT = (K + BK - 1) / BK;

  if constexpr (MODE == 0) {
    // ------------------- synchronous staging (f32 A, K tail) ----------------
    const float* A = (const float*)Aptr;

    auto stage = [&](int k0, bool guard) {
#pragma unroll
      for (int i = 0; i < 2; ++i) {
        int chunk = t + i * 256;
        int r = chunk >> 2;
        int c = (chunk & 3) * 8;
        int gk = k0 + c;
        v8h val = {};
        if (!guard || gk < K) {
          const float* src = &A[(size_t)(m0 + r) * K + gk];
          v4f x0 = *(const v4f*)src;
          v4f x1 = *(const v4f*)(src + 4);
#pragma unroll
          for (int q = 0; q < 4; ++q) { val[q] = (_Float16)x0[q]; val[q + 4] = (_Float16)x1[q]; }
        }
        *(v8h*)&As[r * LDT + c] = val;
      }
#pragma unroll
      for (int i = 0; i < 2; ++i) {
        int chunk = t + i * 256;
        int r = chunk >> 2;
        int c = (chunk & 3) * 8;
        int gk = k0 + c;
        v8h val = {};
        if (!guard || gk < K) val = *(const v8h*)&Wt[(size_t)(n0 + r) * K + gk];
        *(v8h*)&Bs[r * LDT + c] = val;
      }
    };
    auto compute = [&]() {
      v16h afr[2], bfr[4];
#pragma unroll
      for (int i = 0; i < 2; ++i)
        afr[i] = load_frag(&As[(wm * 32 + i * 16 + li) * LDT], kb);
#pragma unroll
      for (int j = 0; j < 4; ++j)
        bfr[j] = load_frag(&Bs[(wn * 64 + j * 16 + li) * LDT], kb);
#pragma unroll
      for (int i = 0; i < 2; ++i)
#pragma unroll
        for (int j = 0; j < 4; ++j)
          acc[i][j] = __builtin_amdgcn_wmma_f32_16x16x32_f16(
              false, afr[i], false, bfr[j], (short)0, acc[i][j], false, false);
    };

    const int KFULL = K / BK;  // full tiles: no bounds checks in the hot loop
    for (int kt = 0; kt < KFULL; ++kt) {
      stage(kt * BK, false);
      __syncthreads();
      compute();
      __syncthreads();
    }
    if (KFULL < KT) {  // guarded K tail (zero-padded)
      stage(KFULL * BK, true);
      __syncthreads();
      compute();
      __syncthreads();
    }
  } else {
    // -------- async double-buffered staging (f16 A, K % 32 == 0) -----------
    const _Float16* A = (const _Float16*)Aptr;
    unsigned long long gA[2], gB[2];
    unsigned int lA[2][2], lB[2][2];  // [buf][chunk]
#pragma unroll
    for (int i = 0; i < 2; ++i) {
      int chunk = t + i * 256;
      int r = chunk >> 2;
      int c = (chunk & 3) * 8;
      gA[i] = (unsigned long long)(uintptr_t)(A + (size_t)(m0 + r) * K + c);
      int gn = n0 + r;
      if (gn > N - 1) gn = N - 1;  // clamp; those acc columns are never stored
      gB[i] = (unsigned long long)(uintptr_t)(Wt + (size_t)gn * K + c);
#pragma unroll
      for (int b = 0; b < 2; ++b) {
        lA[b][i] = lds_addr32(&As[b * BM * LDT + r * LDT + c]);
        lB[b][i] = lds_addr32(&Bs[b * BN * LDT + r * LDT + c]);
      }
    }

    // prologue: tile 0 -> buffer 0
#pragma unroll
    for (int i = 0; i < 2; ++i) async_copy_b128(lA[0][i], gA[i]);
#pragma unroll
    for (int i = 0; i < 2; ++i) async_copy_b128(lB[0][i], gB[i]);

    for (int kt = 0; kt < KT; ++kt) {
      const int cur = kt & 1;
      wait_async0();    // my tile-kt copies have landed in LDS
      __syncthreads();  // everyone's landed; everyone done reading buf[1-cur]
      if (kt + 1 < KT) {
        const unsigned long long koff =
            (unsigned long long)(kt + 1) * (BK * sizeof(_Float16));
#pragma unroll
        for (int i = 0; i < 2; ++i) async_copy_b128(lA[1 - cur][i], gA[i] + koff);
#pragma unroll
        for (int i = 0; i < 2; ++i) async_copy_b128(lB[1 - cur][i], gB[i] + koff);
      }
      const _Float16* Ab = &As[cur * BM * LDT];
      const _Float16* Bb = &Bs[cur * BN * LDT];
      v16h afr[2], bfr[4];
#pragma unroll
      for (int i = 0; i < 2; ++i)
        afr[i] = load_frag(&Ab[(wm * 32 + i * 16 + li) * LDT], kb);
#pragma unroll
      for (int j = 0; j < 4; ++j)
        bfr[j] = load_frag(&Bb[(wn * 64 + j * 16 + li) * LDT], kb);
#pragma unroll
      for (int i = 0; i < 2; ++i)
#pragma unroll
        for (int j = 0; j < 4; ++j)
          acc[i][j] = __builtin_amdgcn_wmma_f32_16x16x32_f16(
              false, afr[i], false, bfr[j], (short)0, acc[i][j], false, false);
    }
  }

  // ---- epilogue: C/D layout: vgpr r, lanes 0-15 => (M=r, N=lane),
  //                lanes 16-31 => (M=8+r, N=lane-16) ----
  const int nlane = lane & 15;
  const int mhalf = (lane >> 4) * 8;
#pragma unroll
  for (int i = 0; i < 2; ++i) {
#pragma unroll
    for (int j = 0; j < 4; ++j) {
      int gn = n0 + wn * 64 + j * 16 + nlane;
      if (gn < N) {
        float bv = bias[gn];
        int gm = m0 + wm * 32 + i * 16 + mhalf;
#pragma unroll
        for (int r = 0; r < 8; ++r) {
          float v = acc[i][j][r] + bv;
          size_t off = (size_t)(gm + r) * N + gn;
          if (MODE == 2) {
            v = 1.0f / (1.0f + __expf(-v));
            ((float*)Out)[off] = v;
          } else {
            v = v > 0.0f ? v : 0.0f;
            ((_Float16*)Out)[off] = (_Float16)v;
          }
        }
      }
    }
  }
}

// ---------------------------------------------------------------------------
// Fused middle: per batch-row,
//   z_e = h2 @ Wez + bez   (256 -> 10)         -> out (f32)
//   VQ  : idx = argmin_v( ||e_v||^2 - 2 z.e_v ), z_q = codebook[idx] -> out
//   d1  = relu(z_q @ Wd1 + bd1) (10 -> 256)    -> ws (f16), decoder input
// ---------------------------------------------------------------------------
__global__ __launch_bounds__(256) void mid_vq_kernel(
    const _Float16* __restrict__ h2,    // [B,256]
    const float* __restrict__ Wez,      // [256,10]
    const float* __restrict__ bez,      // [10]
    const float* __restrict__ codebook, // [64,10]
    const float* __restrict__ Wd1,      // [10,256]
    const float* __restrict__ bd1,      // [256]
    float* __restrict__ ze_out,         // [B,10]
    float* __restrict__ zq_out,         // [B,10]
    _Float16* __restrict__ d1) {        // [B,256]
  __shared__ float sWez[DIM2 * DIMZ];
  __shared__ float sCode[VOCAB * DIMZ];
  __shared__ float sNorm[VOCAB];
  __shared__ float sWd1[DIMZ * DIM2];
  __shared__ float sBd1[DIM2];
  __shared__ float sBez[DIMZ];

  const int t = threadIdx.x;
  for (int i = t; i < DIM2 * DIMZ; i += 256) sWez[i] = Wez[i];
  for (int i = t; i < VOCAB * DIMZ; i += 256) sCode[i] = codebook[i];
  for (int i = t; i < DIMZ * DIM2; i += 256) sWd1[i] = Wd1[i];
  sBd1[t] = bd1[t];
  if (t < DIMZ) sBez[t] = bez[t];
  __syncthreads();
  if (t < VOCAB) {
    float s = 0.0f;
#pragma unroll
    for (int j = 0; j < DIMZ; ++j) { float e = sCode[t * DIMZ + j]; s += e * e; }
    sNorm[t] = s;
  }
  __syncthreads();

  const size_t row = (size_t)blockIdx.x * 256 + t;

  // z_e = h2row @ Wez + bez
  float z[DIMZ];
#pragma unroll
  for (int j = 0; j < DIMZ; ++j) z[j] = sBez[j];
  const _Float16* hrow = &h2[row * DIM2];
  for (int k = 0; k < DIM2; k += 8) {
    v8h hv = *(const v8h*)&hrow[k];
#pragma unroll
    for (int q = 0; q < 8; ++q) {
      float h = (float)hv[q];
      const float* w = &sWez[(k + q) * DIMZ];
#pragma unroll
      for (int j = 0; j < DIMZ; ++j) z[j] += h * w[j];
    }
  }
#pragma unroll
  for (int j = 0; j < DIMZ; ++j) ze_out[row * DIMZ + j] = z[j];

  // VQ argmin (first-min tie break, like jnp.argmin)
  int best = 0;
  float bestd = 3.4e38f;
  for (int v = 0; v < VOCAB; ++v) {
    float d = sNorm[v];
#pragma unroll
    for (int j = 0; j < DIMZ; ++j) d -= 2.0f * z[j] * sCode[v * DIMZ + j];
    if (d < bestd) { bestd = d; best = v; }
  }
  float zq[DIMZ];
#pragma unroll
  for (int j = 0; j < DIMZ; ++j) {
    zq[j] = sCode[best * DIMZ + j];
    zq_out[row * DIMZ + j] = zq[j];
  }

  // d1 = relu(zq @ Wd1 + bd1), f16
  _Float16* drow = &d1[row * DIM2];
  for (int n = 0; n < DIM2; n += 8) {
    v8h o;
#pragma unroll
    for (int q = 0; q < 8; ++q) {
      float a = sBd1[n + q];
#pragma unroll
      for (int j = 0; j < DIMZ; ++j) a += zq[j] * sWd1[j * DIM2 + n + q];
      o[q] = (_Float16)(a > 0.0f ? a : 0.0f);
    }
    *(v8h*)&drow[n] = o;
  }
}

// ---------------------------------------------------------------------------
// Launch: weight prep -> GEMM1 -> GEMM2 -> fused VQ -> GEMM5 -> GEMM6
// ws layout: buf0[B*512 f16] | buf1[B*512 f16] | wt1 | wt2 | wt5 | wt6
// (needs ~137 MB workspace)
// ---------------------------------------------------------------------------
extern "C" void kernel_launch(void* const* d_in, const int* in_sizes, int n_in,
                              void* d_out, int out_size, void* d_ws, size_t ws_size,
                              hipStream_t stream) {
  const float* x   = (const float*)d_in[0];
  const float* We1 = (const float*)d_in[1];
  const float* be1 = (const float*)d_in[2];
  const float* We2 = (const float*)d_in[3];
  const float* be2 = (const float*)d_in[4];
  const float* Wez = (const float*)d_in[5];
  const float* bez = (const float*)d_in[6];
  const float* Wd1 = (const float*)d_in[7];
  const float* bd1 = (const float*)d_in[8];
  const float* Wd2 = (const float*)d_in[9];
  const float* bd2 = (const float*)d_in[10];
  const float* Wd3 = (const float*)d_in[11];
  const float* bd3 = (const float*)d_in[12];
  const float* cb  = (const float*)d_in[13];

  const int B = in_sizes[0] / DIM_IN;  // 65536 (multiple of 128)

  float* out = (float*)d_out;
  float* xp = out;                         // [B,784]
  float* ze = out + (size_t)B * DIM_IN;    // [B,10]
  float* zq = ze + (size_t)B * DIMZ;       // [B,10]

  char* ws = (char*)d_ws;
  _Float16* buf0 = (_Float16*)ws;                              // B*512 halves
  _Float16* buf1 = (_Float16*)(ws + (size_t)B * DIM1 * 2);     // B*512 halves
  _Float16* wt1  = (_Float16*)(ws + (size_t)B * DIM1 * 4);     // [512,784]
  _Float16* wt2  = wt1 + (size_t)DIM1 * DIM_IN;                // [256,512]
  _Float16* wt5  = wt2 + (size_t)DIM2 * DIM1;                  // [512,256]
  _Float16* wt6  = wt5 + (size_t)DIM1 * DIM2;                  // [784,512]

  // ---- weight convert + transpose (tiny) ----
  wconv_kernel<<<(DIM_IN * DIM1 + 255) / 256, 256, 0, stream>>>(We1, wt1, DIM_IN, DIM1);
  wconv_kernel<<<(DIM1 * DIM2 + 255) / 256, 256, 0, stream>>>(We2, wt2, DIM1, DIM2);
  wconv_kernel<<<(DIM2 * DIM1 + 255) / 256, 256, 0, stream>>>(Wd2, wt5, DIM2, DIM1);
  wconv_kernel<<<(DIM1 * DIM_IN + 255) / 256, 256, 0, stream>>>(Wd3, wt6, DIM1, DIM_IN);

  const dim3 blk(256);
  const int MT = B / 128;

  // enc1: relu(x @ We1 + be1)  [B,784]x[784,512] -> buf0 f16
  gemm_wmma_kernel<0><<<dim3(DIM1 / 128, MT), blk, 0, stream>>>(x, wt1, be1, buf0, DIM1, DIM_IN);
  // enc2: relu(h1 @ We2 + be2) [B,512]x[512,256] -> buf1 f16
  gemm_wmma_kernel<1><<<dim3(DIM2 / 128, MT), blk, 0, stream>>>(buf0, wt2, be2, buf1, DIM2, DIM1);
  // fused: z_e -> out, VQ -> z_q -> out, d1 = relu(z_q@Wd1+bd1) -> buf0 f16
  mid_vq_kernel<<<B / 256, 256, 0, stream>>>(buf1, Wez, bez, cb, Wd1, bd1, ze, zq, buf0);
  // dec2: relu(d1 @ Wd2 + bd2) [B,256]x[256,512] -> buf1 f16
  gemm_wmma_kernel<1><<<dim3(DIM1 / 128, MT), blk, 0, stream>>>(buf0, wt5, bd2, buf1, DIM1, DIM2);
  // dec3: sigmoid(d2 @ Wd3 + bd3) [B,512]x[512,784] -> xp f32
  gemm_wmma_kernel<2><<<dim3((DIM_IN + 127) / 128, MT), blk, 0, stream>>>(buf1, wt6, bd3, xp, DIM_IN, DIM1);
}
